// BigramLanguageModel_V2_74801150427876
// MI455X (gfx1250) — compile-verified, matrix operand order checked by reading, not run on hardware
//
#include <hip/hip_runtime.h>

// ---------------------------------------------------------------------------
// BigramLanguageModel forward for MI455X (gfx1250, wave32, WMMA).
//   V=32000, E=768, CTX=2048, B=4, T=2048, M = B*T = 8192
// Pipeline (all matmuls bf16 WMMA with f32 accumulation):
//   1. convert Wq/Wk/Wv/lm_W  f32 -> bf16          (workspace)
//   2. x = tok_emb[idx] + pos_emb   -> bf16        (workspace)
//   3. q,k,v = x @ W    (double-buffered async-LDS WMMA GEMM, bf16 out)
//   4. o = causal-softmax(q k^T / sqrt(E)) v       (flash-attn, WMMA,
//                                                   async K/V staging,
//                                                   S split across 8 waves)
//   5. logits = o @ lm_W + b  (WMMA GEMM, f32 out -> d_out)
//   6. loss = mean NLL via per-row logsumexp       (-> d_out[M*V])
// CDNA5 paths: v_wmma_f32_16x16x32_bf16, global_load_async_to_lds_b128
// (ASYNCcnt DMA, single base + offset: immediates -> no address chains),
// s_wait_asynccnt.
// ---------------------------------------------------------------------------

typedef __attribute__((ext_vector_type(16))) __bf16 v16bf;
typedef __attribute__((ext_vector_type(8)))  float  v8f;

union Frag16 { uint4 q[2]; v16bf v; };   // 32 bytes

static __device__ inline __bf16 to_bf16(float f) {
  unsigned u = __builtin_bit_cast(unsigned, f);
  unsigned r = u + 0x7FFFu + ((u >> 16) & 1u);   // round-to-nearest-even
  return __builtin_bit_cast(__bf16, (unsigned short)(r >> 16));
}
static __device__ inline unsigned pack2(float a, float b) {
  unsigned short ha = __builtin_bit_cast(unsigned short, to_bf16(a));
  unsigned short hb = __builtin_bit_cast(unsigned short, to_bf16(b));
  return (unsigned)ha | ((unsigned)hb << 16);
}

// Async DMA: 16 bytes/lane, global -> LDS, tracked by ASYNCcnt. The 24-bit
// instruction offset is added to BOTH the global and the LDS address
// (ISA 08_async_tensor pseudocode), so a burst of contiguous chunks needs
// only one base address pair. LDS operand = low 32 bits of the generic
// shared pointer (ISA 10.2: LDS aperture uses addr[31:0]).
template <int I, int N>
struct AsyncBurst {
  static __device__ __forceinline__ void run(const void* g, void* l) {
    asm volatile("global_load_async_to_lds_b128 %0, %1, off offset:%2"
                 :
                 : "v"((unsigned)(unsigned long long)l),
                   "v"((unsigned long long)g),
                   "n"(I * 16)
                 : "memory");
    AsyncBurst<I + 1, N>::run(g, l);
  }
};
template <int N>
struct AsyncBurst<N, N> {
  static __device__ __forceinline__ void run(const void*, void*) {}
};
template <int N>
static __device__ __forceinline__ void async_burst(const void* g, void* l) {
  AsyncBurst<0, N>::run(g, l);
}
static __device__ inline void async_copy16(const void* gaddr, void* lds) {
  asm volatile("global_load_async_to_lds_b128 %0, %1, off"
               :
               : "v"((unsigned)(unsigned long long)lds),
                 "v"((unsigned long long)gaddr)
               : "memory");
}
static __device__ inline void wait_async() {
  asm volatile("s_wait_asynccnt 0x0" ::: "memory");
}

// Load a 16x32 bf16 fragment in the CDNA5 WMMA A/B register layout.
// `tile` points at element [0][k0] of a row-major LDS tile with leading
// dimension ldElems. Per ISA 7.12.2: lanes 0-15 hold K 0-7 & 16-23,
// lanes 16-31 hold K 8-15 & 24-31 => two b128 LDS loads per lane.
static __device__ inline v16bf load_frag(const __bf16* tile, int ldElems, int lane) {
  const int m = lane & 15;
  const uint4* p = (const uint4*)(tile + (size_t)m * ldElems);
  const int kb = (lane >> 4) & 1;
  Frag16 f;
  f.q[0] = p[kb];
  f.q[1] = p[kb + 2];
  return f.v;
}

// Same fragment layout gathering a column-major operand: element k of
// lane-column n comes from tile[k*ldElems + n]  (used for B = V in P@V).
static __device__ inline v16bf load_frag_col(const __bf16* tile, int ldElems, int lane) {
  const int n  = lane & 15;
  const int kb = ((lane >> 4) & 1) << 3;   // 0 or 8
  const __bf16* c = tile + n;
  Frag16 f;
  unsigned short* e = (unsigned short*)&f;
#pragma unroll
  for (int i = 0; i < 8; ++i) {
    int k = ((i & 4) << 2) + kb + ((i & 3) << 1);  // {0,16} + kb + 2*(i%4)
    e[2*i]   = __builtin_bit_cast(unsigned short, c[(size_t)k * ldElems]);
    e[2*i+1] = __builtin_bit_cast(unsigned short, c[(size_t)(k+1) * ldElems]);
  }
  return f.v;
}

static __device__ inline v8f zero8() {
  v8f z;
#pragma unroll
  for (int e = 0; e < 8; ++e) z[e] = 0.f;
  return z;
}

// ---------------------------------------------------------------------------
// f32 -> bf16 conversion (2 elements / thread, dword stores)
// ---------------------------------------------------------------------------
__global__ __launch_bounds__(256) void cvt_f32_bf16(const float* __restrict__ s,
                                                    __bf16* __restrict__ d, long n2) {
  long i = (long)blockIdx.x * 256 + threadIdx.x;
  if (i < n2) ((unsigned*)d)[i] = pack2(s[2 * i], s[2 * i + 1]);
}

// ---------------------------------------------------------------------------
// x = tok_emb[idx] + pos_emb   ->  bf16 [M=8192, E=768]
// ---------------------------------------------------------------------------
__global__ __launch_bounds__(256) void embed_kernel(const int* __restrict__ idx,
                                                    const float* __restrict__ tok,
                                                    const float* __restrict__ pos,
                                                    __bf16* __restrict__ x) {
  const int E = 768;
  size_t i = (size_t)blockIdx.x * 256 + threadIdx.x;   // over M * E/2
  if (i >= (size_t)8192 * 384) return;
  int r = (int)(i / 384);
  int e = (int)(i % 384) * 2;
  int t = r & 2047;
  int tk = idx[r];
  float a0 = tok[(size_t)tk * E + e]     + pos[(size_t)t * E + e];
  float a1 = tok[(size_t)tk * E + e + 1] + pos[(size_t)t * E + e + 1];
  ((unsigned*)x)[i] = pack2(a0, a1);
}

// ---------------------------------------------------------------------------
// WMMA GEMM: C[M,N] = A[M,K] * B[K,N]  (A,B bf16 row-major, f32 accumulate)
// Block: 256 threads = 8 waves, tile 128x128, K-step 32, double-buffered.
// A tile staged by async-LDS DMA; B tile needs a bf16 transpose so it goes
// global->VGPR->LDS, with its global loads issued before the WMMAs and its
// LDS stores landing in the other buffer after them -> one barrier per step.
// F32OUT=1: write f32 + bias (logits). F32OUT=0: write bf16.
// M, N multiples of 128; K multiple of 32.
// ---------------------------------------------------------------------------
template <bool F32OUT>
__global__ __launch_bounds__(256) void gemm_bf16_wmma(const __bf16* __restrict__ A,
                                                      const __bf16* __restrict__ B,
                                                      const float* __restrict__ bias,
                                                      float* __restrict__ outF,
                                                      __bf16* __restrict__ outH,
                                                      int M, int N, int K) {
  __shared__ alignas(16) __bf16 sA[2][128][32];
  __shared__ alignas(16) __bf16 sB[2][128][32];   // [n][k]

  const int tid  = threadIdx.x;
  const int lane = tid & 31;
  const int wave = tid >> 5;
  const int wm   = wave >> 2;
  const int wn   = wave & 3;
  const int m0   = blockIdx.x * 128;
  const int n0   = blockIdx.y * 128;

  // A: 512 16-byte chunks; this thread covers chunks tid and tid+256.
  const int arow = tid >> 2;          // 0..63
  const int acol = (tid & 3) * 8;     // bf16 elements
  unsigned breg[8];

  auto stage_A = [&](int k0, int buf) {
#pragma unroll
    for (int h = 0; h < 2; ++h) {
      int row = arow + h * 64;
      async_copy16(A + (size_t)(m0 + row) * K + k0 + acol, &sA[buf][row][acol]);
    }
  };
  auto load_B = [&](int k0) {
#pragma unroll
    for (int i = 0; i < 8; ++i) {
      int d = tid + i * 256;
      int kk = d >> 6, dn = d & 63;
      breg[i] = ((const unsigned*)(B + (size_t)(k0 + kk) * N + n0))[dn];
    }
  };
  auto store_B = [&](int buf) {
#pragma unroll
    for (int i = 0; i < 8; ++i) {
      int d = tid + i * 256;
      int kk = d >> 6, dn = d & 63;
      sB[buf][2 * dn][kk]     = __builtin_bit_cast(__bf16, (unsigned short)(breg[i] & 0xffffu));
      sB[buf][2 * dn + 1][kk] = __builtin_bit_cast(__bf16, (unsigned short)(breg[i] >> 16));
    }
  };

  v8f acc[4][2];
#pragma unroll
  for (int i = 0; i < 4; ++i)
#pragma unroll
    for (int j = 0; j < 2; ++j) acc[i][j] = zero8();

  // Prologue: tile 0 into buffer 0.
  stage_A(0, 0);
  load_B(0);
  store_B(0);
  wait_async();
  __syncthreads();

  int cur = 0;
  for (int k0 = 0; k0 < K; k0 += 32) {
    const bool more = (k0 + 32) < K;
    if (more) {                     // prefetch next tile while we compute
      stage_A(k0 + 32, cur ^ 1);
      load_B(k0 + 32);
    }

    v16bf bf0 = load_frag(&sB[cur][wn * 32][0],      32, lane);
    v16bf bf1 = load_frag(&sB[cur][wn * 32 + 16][0], 32, lane);
#pragma unroll
    for (int i = 0; i < 4; ++i) {
      v16bf af = load_frag(&sA[cur][wm * 64 + i * 16][0], 32, lane);
      acc[i][0] = __builtin_amdgcn_wmma_f32_16x16x32_bf16(
          false, af, false, bf0, (short)0, acc[i][0], false, false);
      acc[i][1] = __builtin_amdgcn_wmma_f32_16x16x32_bf16(
          false, af, false, bf1, (short)0, acc[i][1], false, false);
    }

    if (more) store_B(cur ^ 1);     // into the other buffer, no race
    wait_async();
    __syncthreads();
    cur ^= 1;
  }

  // Epilogue. C layout: VGPR r, lanes 0-15 -> row r, lanes 16-31 -> row r+8.
  const int lcol  = lane & 15;
  const int rhalf = (lane >> 4) << 3;
#pragma unroll
  for (int j = 0; j < 2; ++j) {
    const int col = n0 + wn * 32 + j * 16 + lcol;
    float bv = 0.f;
    if constexpr (F32OUT) bv = bias[col];
#pragma unroll
    for (int i = 0; i < 4; ++i) {
#pragma unroll
      for (int r = 0; r < 8; ++r) {
        const int row = m0 + wm * 64 + i * 16 + rhalf + r;
        if constexpr (F32OUT) outF[(size_t)row * N + col] = acc[i][j][r] + bv;
        else                  outH[(size_t)row * N + col] = to_bf16(acc[i][j][r]);
      }
    }
  }
}

// ---------------------------------------------------------------------------
// Flash attention, single head, causal. head_size == E == 768, T == 2048.
// Block = 256 threads (8 waves) handles one (batch, 16-row query tile).
// Key loop in steps of 32; K then V staged through one shared 32x768 buffer
// via async-LDS DMA bursts. S = Q K^T split across the 8 waves (each wave
// computes a 96-wide K-slice partial; partials summed through LDS). Each
// wave owns 96 output columns (6 accumulators) for P@V.
// LDS: 24 (Q) + 48 (KV) + 8 (P) + 16 (S partials) = 96 KB.
// ---------------------------------------------------------------------------
__global__ __launch_bounds__(256) void attn_flash(const __bf16* __restrict__ q,
                                                  const __bf16* __restrict__ k,
                                                  const __bf16* __restrict__ v,
                                                  __bf16* __restrict__ o,
                                                  float scale) {
  constexpr int E = 768, T = 2048;
  __shared__ alignas(16) __bf16 sQ[16][E];      // 24 KB
  __shared__ alignas(16) __bf16 sKV[32][E];     // 48 KB (K, then V)
  __shared__ alignas(16) __bf16 sP[8][16][32];  // 8 KB  per-wave P tile
  __shared__ alignas(16) float  sSp[8][16][32]; // 16 KB per-wave S partials

  const int tid  = threadIdx.x;
  const int lane = tid & 31;
  const int wave = tid >> 5;
  const int b    = blockIdx.x >> 7;     // T/16 = 128 query tiles per batch
  const int qt   = blockIdx.x & 127;
  const int q0   = qt * 16;
  const size_t base = (size_t)b * T * E;

  // Async-stage Q: 16 rows x 96 chunks(16B); 16 threads/row, 6 chunks each.
  {
    const int row = tid >> 4;
    const int c0  = (tid & 15) * 6;
    async_burst<6>(q + base + (size_t)(q0 + row) * E + c0 * 8, &sQ[row][c0 * 8]);
  }

  // Async-stage 32 rows x 768 into sKV: 8 threads/row, 12 chunks each.
  const int kvrow = tid >> 3;
  const int kvc0  = (tid & 7) * 12;
  auto stage_kv = [&](const __bf16* src) {
    async_burst<12>(src + (size_t)kvrow * E + kvc0 * 8, &sKV[kvrow][kvc0 * 8]);
  };

  v8f acc[6];
#pragma unroll
  for (int nt = 0; nt < 6; ++nt) acc[nt] = zero8();
  float rm[8], rl[8];
#pragma unroll
  for (int r = 0; r < 8; ++r) { rm[r] = -3.0e38f; rl[r] = 0.f; }

  const int lcol  = lane & 15;
  const int rhalf = (lane >> 4) << 3;
  const int cbase = wave * 96;

  for (int j0 = 0; j0 < q0 + 16; j0 += 32) {
    __syncthreads();                     // prior P@V reads of sKV complete
    stage_kv(k + base + (size_t)j0 * E); // also covers the Q prologue DMA
    wait_async();
    __syncthreads();

    // Partial S[16x32] over this wave's 96-wide K slice (3 steps, 6 WMMA).
    v8f s0 = zero8(), s1 = zero8();
    for (int kk = wave * 96; kk < wave * 96 + 96; kk += 32) {
      v16bf aq = load_frag(&sQ[0][kk],   E, lane);
      v16bf b0 = load_frag(&sKV[0][kk],  E, lane);
      v16bf b1 = load_frag(&sKV[16][kk], E, lane);
      s0 = __builtin_amdgcn_wmma_f32_16x16x32_bf16(false, aq, false, b0,
                                                   (short)0, s0, false, false);
      s1 = __builtin_amdgcn_wmma_f32_16x16x32_bf16(false, aq, false, b1,
                                                   (short)0, s1, false, false);
    }
#pragma unroll
    for (int r = 0; r < 8; ++r) {        // store partials in C-layout positions
      sSp[wave][rhalf + r][lcol]      = s0[r];
      sSp[wave][rhalf + r][16 + lcol] = s1[r];
    }
    __syncthreads();                     // all waves done with K & partials

    stage_kv(v + base + (size_t)j0 * E); // overlap V DMA with softmax below

    // Sum 8 partials + online softmax. Row (rhalf+r) lives in vgpr r across
    // this lane-half; shfl_xor over masks 1,2,4,8 stays inside the half.
    float corr[8];
#pragma unroll
    for (int r = 0; r < 8; ++r) {
      const int row = q0 + rhalf + r;
      float x0 = 0.f, x1 = 0.f;
#pragma unroll
      for (int w = 0; w < 8; ++w) {
        x0 += sSp[w][rhalf + r][lcol];
        x1 += sSp[w][rhalf + r][16 + lcol];
      }
      x0 *= scale; if (j0 + lcol > row)      x0 = -1e30f;
      x1 *= scale; if (j0 + 16 + lcol > row) x1 = -1e30f;
      float mx = fmaxf(x0, x1);
#pragma unroll
      for (int off = 1; off < 16; off <<= 1) mx = fmaxf(mx, __shfl_xor(mx, off, 32));
      float mnew = fmaxf(rm[r], mx);
      float c  = __expf(rm[r] - mnew);
      float p0 = __expf(x0 - mnew);
      float p1 = __expf(x1 - mnew);
      float ps = p0 + p1;
#pragma unroll
      for (int off = 1; off < 16; off <<= 1) ps += __shfl_xor(ps, off, 32);
      rl[r] = rl[r] * c + ps;
      rm[r] = mnew;
      corr[r] = c;
      sP[wave][rhalf + r][lcol]      = to_bf16(p0);   // wave-local, in-order LDS
      sP[wave][rhalf + r][16 + lcol] = to_bf16(p1);
    }
#pragma unroll
    for (int nt = 0; nt < 6; ++nt)
#pragma unroll
      for (int r = 0; r < 8; ++r) acc[nt][r] *= corr[r];

    wait_async();
    __syncthreads();                     // V resident in sKV

    // O[16x96 chunk] += P[16x32] @ V[32x96 chunk]
    v16bf pf = load_frag(&sP[wave][0][0], 32, lane);
#pragma unroll
    for (int nt = 0; nt < 6; ++nt) {
      v16bf bv = load_frag_col(&sKV[0][cbase + nt * 16], E, lane);
      acc[nt] = __builtin_amdgcn_wmma_f32_16x16x32_bf16(false, pf, false, bv,
                                                        (short)0, acc[nt], false, false);
    }
  }

#pragma unroll
  for (int nt = 0; nt < 6; ++nt) {
#pragma unroll
    for (int r = 0; r < 8; ++r) {
      size_t off = base + (size_t)(q0 + rhalf + r) * E + cbase + nt * 16 + lcol;
      o[off] = to_bf16(acc[nt][r] / rl[r]);
    }
  }
}

// ---------------------------------------------------------------------------
// Loss: per row of logits, nll = logsumexp(row) - row[target]; mean -> *loss
// Row = 128 KB; second pass re-reads from the 192 MB L2. float4 streams.
// ---------------------------------------------------------------------------
__global__ void zero_loss(float* loss) {
  if (threadIdx.x == 0 && blockIdx.x == 0) *loss = 0.f;
}

__global__ __launch_bounds__(256) void loss_kernel(const float* __restrict__ logits,
                                                   const int* __restrict__ target,
                                                   float* __restrict__ loss) {
  __shared__ float red[256];
  const int V4 = 32000 / 4;
  const int row = blockIdx.x;
  const float4* p4 = (const float4*)(logits + (size_t)row * 32000);
  const int tid = threadIdx.x;

  float mx = -3.4e38f;
  for (int i = tid; i < V4; i += 256) {
    float4 x = p4[i];
    mx = fmaxf(fmaxf(fmaxf(mx, x.x), x.y), fmaxf(x.z, x.w));
  }
  red[tid] = mx;
  __syncthreads();
  for (int s = 128; s > 0; s >>= 1) {
    if (tid < s) red[tid] = fmaxf(red[tid], red[tid + s]);
    __syncthreads();
  }
  mx = red[0];
  __syncthreads();

  float sum = 0.f;
  for (int i = tid; i < V4; i += 256) {
    float4 x = p4[i];
    sum += __expf(x.x - mx) + __expf(x.y - mx) + __expf(x.z - mx) + __expf(x.w - mx);
  }
  red[tid] = sum;
  __syncthreads();
  for (int s = 128; s > 0; s >>= 1) {
    if (tid < s) red[tid] += red[tid + s];
    __syncthreads();
  }
  if (tid == 0) {
    float tv = ((const float*)p4)[target[row]];
    float nll = logf(red[0]) + mx - tv;
    atomicAdd(loss, nll * (1.0f / 8192.0f));
  }
}

// ---------------------------------------------------------------------------
extern "C" void kernel_launch(void* const* d_in, const int* in_sizes, int n_in,
                              void* d_out, int out_size, void* d_ws, size_t ws_size,
                              hipStream_t stream) {
  (void)in_sizes; (void)n_in; (void)out_size; (void)ws_size;
  const int B = 4, T = 2048, E = 768, V = 32000;
  const int M = B * T;

  const int*   idx     = (const int*)d_in[0];
  const int*   target  = (const int*)d_in[1];
  const float* tok_emb = (const float*)d_in[2];
  const float* pos_emb = (const float*)d_in[3];
  const float* Wq      = (const float*)d_in[4];
  const float* Wk      = (const float*)d_in[5];
  const float* Wv      = (const float*)d_in[6];
  const float* lmW     = (const float*)d_in[7];
  const float* lmb     = (const float*)d_in[8];

  float* logits = (float*)d_out;                 // [M, V] f32
  float* loss   = logits + (size_t)M * V;        // scalar

  // Workspace layout (~116 MB total)
  char* cur = (char*)d_ws;
  auto alloc = [&](size_t bytes) {
    char* p = cur;
    cur += (bytes + 255) & ~(size_t)255;
    return p;
  };
  __bf16* xh   = (__bf16*)alloc((size_t)M * E * 2);
  __bf16* qh   = (__bf16*)alloc((size_t)M * E * 2);
  __bf16* kh   = (__bf16*)alloc((size_t)M * E * 2);
  __bf16* vh   = (__bf16*)alloc((size_t)M * E * 2);
  __bf16* oh   = (__bf16*)alloc((size_t)M * E * 2);
  __bf16* wqh  = (__bf16*)alloc((size_t)E * E * 2);
  __bf16* wkh  = (__bf16*)alloc((size_t)E * E * 2);
  __bf16* wvh  = (__bf16*)alloc((size_t)E * E * 2);
  __bf16* lmWh = (__bf16*)alloc((size_t)E * V * 2);

  // 1. weight conversions
  {
    long n2 = (long)E * E / 2;
    int g = (int)((n2 + 255) / 256);
    cvt_f32_bf16<<<g, 256, 0, stream>>>(Wq, wqh, n2);
    cvt_f32_bf16<<<g, 256, 0, stream>>>(Wk, wkh, n2);
    cvt_f32_bf16<<<g, 256, 0, stream>>>(Wv, wvh, n2);
    long n2w = (long)E * V / 2;
    cvt_f32_bf16<<<(int)((n2w + 255) / 256), 256, 0, stream>>>(lmW, lmWh, n2w);
  }

  // 2. embedding gather + add
  embed_kernel<<<(M * (E / 2) + 255) / 256, 256, 0, stream>>>(idx, tok_emb, pos_emb, xh);

  // 3. q, k, v projections (bf16 out)
  {
    dim3 grid(M / 128, E / 128);
    gemm_bf16_wmma<false><<<grid, 256, 0, stream>>>(xh, wqh, nullptr, nullptr, qh, M, E, E);
    gemm_bf16_wmma<false><<<grid, 256, 0, stream>>>(xh, wkh, nullptr, nullptr, kh, M, E, E);
    gemm_bf16_wmma<false><<<grid, 256, 0, stream>>>(xh, wvh, nullptr, nullptr, vh, M, E, E);
  }

  // 4. causal attention
  {
    float scale = 1.0f / sqrtf((float)E);
    attn_flash<<<B * (T / 16), 256, 0, stream>>>(qh, kh, vh, oh, scale);
  }

  // 5. logits = o @ lm_W + b  (f32 to d_out)
  {
    dim3 grid(M / 128, V / 128);
    gemm_bf16_wmma<true><<<grid, 256, 0, stream>>>(oh, lmWh, lmb, logits, nullptr, M, V, E);
  }

  // 6. loss
  zero_loss<<<1, 1, 0, stream>>>(loss);
  loss_kernel<<<M, 256, 0, stream>>>(logits, target, loss);
}
